// Attention_37288906064386
// MI455X (gfx1250) — matrix operand order repaired and checked by reading, not compiled, not run on hardware
//
#include <hip/hip_runtime.h>
#include <hip/hip_bf16.h>
#include <stdint.h>

typedef __attribute__((ext_vector_type(16))) _Float16 v16h;
typedef __attribute__((ext_vector_type(8)))  _Float16 v8h;
typedef __attribute__((ext_vector_type(8)))  float    v8f;

#define B_   4
#define S_   1024
#define D_   2048
#define H_   32
#define KVH_ 8
#define HD_  64
#define QKVN 3072   // H*HD + 2*KVH*HD

__device__ __forceinline__ v16h cat8(v8h lo, v8h hi) {
    return __builtin_shufflevector(lo, hi, 0,1,2,3,4,5,6,7,8,9,10,11,12,13,14,15);
}
__device__ __forceinline__ unsigned pk2(float a, float b) {
    union { _Float16 h[2]; unsigned u; } t;
    t.h[0] = (_Float16)a; t.h[1] = (_Float16)b; return t.u;
}
// lane i <- lane (i ^ 16), single VALU op (v_permlanex16_b32), no LDS round trip
__device__ __forceinline__ unsigned xor16u(unsigned v) {
    return (unsigned)__builtin_amdgcn_permlanex16((int)v, (int)v, 0x76543210, 0xfedcba98, false, false);
}
__device__ __forceinline__ float xor16f(float v) {
    return __int_as_float(__builtin_amdgcn_permlanex16(__float_as_int(v), __float_as_int(v),
                                                       0x76543210, 0xfedcba98, false, false));
}

// ---------------- elementwise prep kernels ----------------

__global__ void xconv_kernel(const float* __restrict__ x, _Float16* __restrict__ xh, int n) {
    int i = blockIdx.x * 256 + threadIdx.x;
    if (i < n) xh[i] = (_Float16)x[i];
}

// WqkvT[n][k] = W*(k, n-part), n in [0,3072), k in [0,2048). k fastest -> coalesced writes.
__global__ void wqkv_trans_kernel(const float* __restrict__ Wq, const float* __restrict__ Wk,
                                  const float* __restrict__ Wv, _Float16* __restrict__ Wt) {
    size_t t = (size_t)blockIdx.x * 256 + threadIdx.x;  // 3072*2048
    int k = (int)(t & 2047);
    int n = (int)(t >> 11);
    float v;
    if (n < 2048)      v = Wq[(size_t)k * 2048 + n];
    else if (n < 2560) v = Wk[(size_t)k * 512 + (n - 2048)];
    else               v = Wv[(size_t)k * 512 + (n - 2560)];
    Wt[t] = (_Float16)v;
}

__global__ void wo_trans_kernel(const float* __restrict__ Wo, _Float16* __restrict__ Wt) {
    size_t t = (size_t)blockIdx.x * 256 + threadIdx.x;  // 2048*2048
    int k = (int)(t & 2047);
    int n = (int)(t >> 11);
    Wt[t] = (_Float16)Wo[(size_t)k * 2048 + n];
}

// RoPE on Q and K sections of qkv (f32), write f16. Q pre-scaled by 1/sqrt(HD)=0.125.
__global__ void rope_kernel(const float* __restrict__ qkv, const float* __restrict__ cosT,
                            const float* __restrict__ sinT,
                            _Float16* __restrict__ qh, _Float16* __restrict__ kh) {
    size_t t = (size_t)blockIdx.x * 256 + threadIdx.x;  // B*S*40*64
    int d = (int)(t & 63);
    int j = (int)((t >> 6) % 40);          // 0..31 Q heads, 32..39 K heads
    size_t bs = t / (64 * 40);             // 0..B*S-1
    int s = (int)(bs & (S_ - 1));
    int b = (int)(bs >> 10);
    int col  = (j < 32) ? (j * 64 + d) : (2048 + (j - 32) * 64 + d);
    int pcol = (d < 32) ? col + 32 : col - 32;
    float v  = qkv[bs * QKVN + col];
    float pv = qkv[bs * QKVN + pcol];
    float rv = (d < 32) ? -pv : pv;
    float out = v * cosT[s * 64 + d] + rv * sinT[s * 64 + d];
    if (j < 32) qh[(((size_t)(b * H_ + j)   * S_) + s) * HD_ + d] = (_Float16)(out * 0.125f);
    else        kh[(((size_t)(b * KVH_ + (j - 32)) * S_) + s) * HD_ + d] = (_Float16)out;
}

// V^T: vT[b][g][d][s] = qkv[b*S+s][2560 + g*64 + d]  (f32 -> f16)
__global__ void vtrans_kernel(const float* __restrict__ qkv, _Float16* __restrict__ vT) {
    size_t t = (size_t)blockIdx.x * 256 + threadIdx.x;  // B*KVH*64*1024, s fastest
    int s = (int)(t & 1023);
    int d = (int)((t >> 10) & 63);
    int g = (int)((t >> 16) & 7);
    int b = (int)(t >> 19);
    vT[t] = (_Float16)qkv[((size_t)(b * S_ + s)) * QKVN + 2560 + g * 64 + d];
}

// ---------------- WMMA GEMM: C[M,N] f32 = A[M,K] f16 @ Bt[N,K]^T f16 ----------------
// Block 128x128, 8 waves (2x4), wave tile 64x32 = 4x2 WMMA C tiles. K step 32.
// Double-buffered LDS + register prefetch; N,K compile-time for immediate-offset stores.

template <int N, int K>
__global__ __launch_bounds__(256)
void gemm_f16_kernel(const _Float16* __restrict__ A, const _Float16* __restrict__ Bt,
                     float* __restrict__ C) {
    __shared__ _Float16 lA[2][128 * 40];
    __shared__ _Float16 lB[2][128 * 40];
    const int tid  = threadIdx.x;
    const int lane = tid & 31;
    const int wave = tid >> 5;
    const int wr = wave >> 2;       // 0..1 (row group of 64)
    const int wc = wave & 3;        // 0..3 (col group of 32)
    const int ln = lane & 15;
    const int hl = lane >> 4;
    const int rowBase = blockIdx.y * 128;
    const int colBase = blockIdx.x * 128;

    // staging assignment: thread covers 16B chunks c0=tid and c1=tid+256 of the 128x32 tiles
    const int r0 = tid >> 2,         kc0 = (tid & 3) * 8;
    const int r1 = (tid + 256) >> 2, kc1 = (tid & 3) * 8;   // (tid+256)&3 == tid&3
    const _Float16* gA0 = A  + (size_t)(rowBase + r0) * K + kc0;
    const _Float16* gA1 = A  + (size_t)(rowBase + r1) * K + kc1;
    const _Float16* gB0 = Bt + (size_t)(colBase + r0) * K + kc0;
    const _Float16* gB1 = Bt + (size_t)(colBase + r1) * K + kc1;
    const int sA0 = r0 * 40 + kc0;
    const int sA1 = r1 * 40 + kc1;

    // loop-invariant fragment LDS offsets (halves)
    int aoff[4], boff[2];
#pragma unroll
    for (int rt = 0; rt < 4; ++rt) aoff[rt] = (wr * 64 + rt * 16 + ln) * 40 + 8 * hl;
#pragma unroll
    for (int ct = 0; ct < 2; ++ct) boff[ct] = (wc * 32 + ct * 16 + ln) * 40 + 8 * hl;

    v8f acc[4][2];
    v8f zero = {};
#pragma unroll
    for (int rt = 0; rt < 4; ++rt)
#pragma unroll
        for (int ct = 0; ct < 2; ++ct) acc[rt][ct] = zero;

    // prefetch first k-slice into registers
    v8h ra0 = *(const v8h*)gA0, ra1 = *(const v8h*)gA1;
    v8h rb0 = *(const v8h*)gB0, rb1 = *(const v8h*)gB1;

    int p = 0;
    for (int k0 = 0; k0 < K; k0 += 32) {
        *(v8h*)&lA[p][sA0] = ra0;
        *(v8h*)&lA[p][sA1] = ra1;
        *(v8h*)&lB[p][sA0] = rb0;
        *(v8h*)&lB[p][sA1] = rb1;
        if (k0 + 32 < K) {                      // prefetch next slice during compute
            ra0 = *(const v8h*)(gA0 + k0 + 32);
            ra1 = *(const v8h*)(gA1 + k0 + 32);
            rb0 = *(const v8h*)(gB0 + k0 + 32);
            rb1 = *(const v8h*)(gB1 + k0 + 32);
        }
        __syncthreads();

        v16h af[4], bf[2];
#pragma unroll
        for (int rt = 0; rt < 4; ++rt) {
            v8h lo = *(const v8h*)&lA[p][aoff[rt]];
            v8h hi = *(const v8h*)&lA[p][aoff[rt] + 16];
            af[rt] = cat8(lo, hi);
        }
#pragma unroll
        for (int ct = 0; ct < 2; ++ct) {
            v8h lo = *(const v8h*)&lB[p][boff[ct]];
            v8h hi = *(const v8h*)&lB[p][boff[ct] + 16];
            bf[ct] = cat8(lo, hi);
        }
#pragma unroll
        for (int rt = 0; rt < 4; ++rt)
#pragma unroll
            for (int ct = 0; ct < 2; ++ct)
                acc[rt][ct] = __builtin_amdgcn_wmma_f32_16x16x32_f16(
                    false, af[rt], false, bf[ct], (short)0, acc[rt][ct], false, false);
        p ^= 1;
    }

    // epilogue: one base address, all offsets compile-time immediates
    float* cp = C + (size_t)(rowBase + wr * 64 + hl * 8) * N + colBase + wc * 32 + ln;
#pragma unroll
    for (int rt = 0; rt < 4; ++rt)
#pragma unroll
        for (int ct = 0; ct < 2; ++ct)
#pragma unroll
            for (int r = 0; r < 8; ++r)
                cp[(rt * 16 + r) * N + ct * 16] = acc[rt][ct][r];
}

// ---------------- Attention: one wave per 16-query tile, transposed-scores flash ----------------

__global__ __launch_bounds__(128)
void attn_kernel(const _Float16* __restrict__ qh, const _Float16* __restrict__ kh,
                 const _Float16* __restrict__ vT, _Float16* __restrict__ attn) {
    const int wid  = blockIdx.x * 4 + (threadIdx.x >> 5);  // 0..8191
    const int qt   = wid & 63;
    const int h    = (wid >> 6) & 31;
    const int b    = wid >> 11;
    const int g    = h >> 2;
    const int lane = threadIdx.x & 31;
    const int ln   = lane & 15;
    const int hl   = lane >> 4;

    // Q as B-operand of S^T = K * Q^T : lane ln holds query row qt*16+ln
    const _Float16* qrow = qh + (((size_t)(b * H_ + h) * S_) + qt * 16 + ln) * HD_;
    const v16h qb0 = *(const v16h*)(qrow + 16 * hl);        // hd 16hl..16hl+15
    const v16h qb1 = *(const v16h*)(qrow + 32 + 16 * hl);   // hd 32+16hl..

    const _Float16* kbase = kh + ((size_t)(b * KVH_ + g) * S_) * HD_;
    const _Float16* vbase = vT + ((size_t)(b * KVH_ + g) * HD_) * S_;

    v8f zero = {};
    v8f o[4];
#pragma unroll
    for (int t = 0; t < 4; ++t) o[t] = zero;
    float m = -1e30f, l = 0.f;

    for (int kc = 0; kc < S_; kc += 32) {
        // S^T tiles: (16 keys x 16 queries), contraction over HD in 2 chunks of 32
        v8f s[2];
#pragma unroll
        for (int t = 0; t < 2; ++t) {
            const _Float16* krow = kbase + (size_t)(kc + t * 16 + ln) * HD_;
            v16h a0 = cat8(*(const v8h*)(krow + 16 * hl),      *(const v8h*)(krow + 8 + 16 * hl));
            v16h a1 = cat8(*(const v8h*)(krow + 32 + 16 * hl), *(const v8h*)(krow + 40 + 16 * hl));
            v8f c = __builtin_amdgcn_wmma_f32_16x16x32_f16(false, a0, false, qb0, (short)0, zero, false, false);
            c     = __builtin_amdgcn_wmma_f32_16x16x32_f16(false, a1, false, qb1, (short)0, c,    false, false);
            s[t] = c;
        }
        // online softmax over keys (rows of S^T): per-lane 16 keys + one cross-half permlane
        float tmax = -1e30f;
#pragma unroll
        for (int r = 0; r < 8; ++r) {
            tmax = fmaxf(tmax, s[0][r]);
            tmax = fmaxf(tmax, s[1][r]);
        }
        tmax = fmaxf(tmax, xor16f(tmax));
        float mn  = fmaxf(m, tmax);
        float fac = __expf(m - mn);
        float p0[8], p1[8];
        float rs = 0.f;
#pragma unroll
        for (int r = 0; r < 8; ++r) {
            p0[r] = __expf(s[0][r] - mn);
            p1[r] = __expf(s[1][r] - mn);
            rs += p0[r] + p1[r];
        }
        rs += xor16f(rs);
        l = l * fac + rs;
        m = mn;
#pragma unroll
        for (int t = 0; t < 4; ++t) o[t] = o[t] * fac;

        // Build P^T as B-operand (32 keys x 16 queries): lane half hl needs keys 16*hl..16*hl+15
        union { unsigned u[8]; v16h h; } pb;
#pragma unroll
        for (int j = 0; j < 4; ++j) {
            unsigned u0 = pk2(p0[2 * j], p0[2 * j + 1]);   // own keys 8*hl + {2j,2j+1}
            unsigned u1 = pk2(p1[2 * j], p1[2 * j + 1]);   // own keys 16+8*hl + {2j,2j+1}
            unsigned y0 = xor16u(u0);
            unsigned y1 = xor16u(u1);
            pb.u[j]     = hl ? y1 : u0;   // keys 16*hl + {2j,2j+1}
            pb.u[4 + j] = hl ? u1 : y0;   // keys 16*hl + 8 + {2j,2j+1}
        }
        // O^T += V^T * P^T  : 4 hd tiles of 16
#pragma unroll
        for (int t = 0; t < 4; ++t) {
            const _Float16* vrow = vbase + (size_t)(t * 16 + ln) * S_ + kc;
            v16h va = cat8(*(const v8h*)(vrow + 16 * hl), *(const v8h*)(vrow + 8 + 16 * hl));
            o[t] = __builtin_amdgcn_wmma_f32_16x16x32_f16(false, va, false, pb.h, (short)0, o[t], false, false);
        }
    }

    const float inv = 1.0f / l;
    const size_t q = (size_t)qt * 16 + ln;
    _Float16* obase = attn + ((size_t)(b * S_) + q) * (H_ * HD_) + h * HD_;
#pragma unroll
    for (int t = 0; t < 4; ++t) {
        union { unsigned u[4]; v8h h; } pk;
#pragma unroll
        for (int j = 0; j < 4; ++j)
            pk.u[j] = pk2(o[t][2 * j] * inv, o[t][2 * j + 1] * inv);
        *(v8h*)(obase + t * 16 + 8 * hl) = pk.h;   // hd = 16t + 8*hl + r
    }
}

// ---------------- launch ----------------

extern "C" void kernel_launch(void* const* d_in, const int* in_sizes, int n_in,
                              void* d_out, int out_size, void* d_ws, size_t ws_size,
                              hipStream_t stream) {
    const float* x    = (const float*)d_in[0];
    const float* cosT = (const float*)d_in[1];
    const float* sinT = (const float*)d_in[2];
    const float* Wq   = (const float*)d_in[3];
    const float* Wk   = (const float*)d_in[4];
    const float* Wv   = (const float*)d_in[5];
    const float* Wo   = (const float*)d_in[6];
    float* out = (float*)d_out;

    char* ws = (char*)d_ws;
    _Float16* xh    = (_Float16*)(ws);                       // 16.78 MB (reused as attn out)
    _Float16* wqkvT = (_Float16*)(ws + 16777216);            // 12.58 MB
    _Float16* woT   = (_Float16*)(ws + 29360128);            //  8.39 MB
    float*    qkv   = (float*)   (ws + 37748736);            // 50.33 MB
    _Float16* qh    = (_Float16*)(ws + 88080384);            // 16.78 MB
    _Float16* kh    = (_Float16*)(ws + 104857600);           //  4.19 MB
    _Float16* vTh   = (_Float16*)(ws + 109051904);           //  4.19 MB
    _Float16* attnh = xh;                                    // alias: xh dead after gemm1

    // 1. convert x -> f16
    xconv_kernel<<<32768, 256, 0, stream>>>(x, xh, B_ * S_ * D_);
    // 2. transpose weights -> f16
    wqkv_trans_kernel<<<24576, 256, 0, stream>>>(Wq, Wk, Wv, wqkvT);
    wo_trans_kernel<<<16384, 256, 0, stream>>>(Wo, woT);
    // 3. QKV projection
    gemm_f16_kernel<QKVN, D_><<<dim3(QKVN / 128, (B_ * S_) / 128), 256, 0, stream>>>(
        xh, wqkvT, qkv);
    // 4. RoPE on Q,K (Q pre-scaled), V transpose
    rope_kernel<<<40960, 256, 0, stream>>>(qkv, cosT, sinT, qh, kh);
    vtrans_kernel<<<8192, 256, 0, stream>>>(qkv, vTh);
    // 5. attention (one wave per 16-query tile; 8192 tiles / 4 waves per block)
    attn_kernel<<<2048, 128, 0, stream>>>(qh, kh, vTh, attnh);
    // 6. output projection -> f32 out
    gemm_f16_kernel<D_, D_><<<dim3(D_ / 128, (B_ * S_) / 128), 256, 0, stream>>>(
        attnh, woT, out);
}
